// BiMultiHeadAttention_43301860278290
// MI455X (gfx1250) — matrix operand order repaired and checked by reading, not compiled, not run on hardware
//
#include <hip/hip_runtime.h>
#include <hip/hip_fp16.h>
#include <math.h>

typedef __attribute__((ext_vector_type(16))) _Float16 v16h;
typedef __attribute__((ext_vector_type(8)))  _Float16 v8h;
typedef __attribute__((ext_vector_type(8)))  float    v8f;
typedef __attribute__((ext_vector_type(4)))  float    v4f;

#define B_   2
#define TV_  16000
#define TL_  256
#define EMB_ 256
#define H_   8
#define D_   32
#define MAXC 50000.0f
#define NEGM (-9e15f)

// ---------------------------------------------------------------- WMMA core
__device__ inline v8f wmma_f16(v16h a, v16h b, v8f c) {
  // D = A(16x32 f16) * B(32x16 f16) + C(16x16 f32)
  return __builtin_amdgcn_wmma_f32_16x16x32_f16(false, a, false, b,
                                                (short)0, c, false, false);
}

// A-fragment (16x32 f16, row-major, M = lane&15). ISA layout: per lane the 16
// K-values are two contiguous 8-half (16B) runs at k = hf*8 and k = 16+hf*8.
__device__ inline v16h frag_a_f16_vec(const _Float16* s, int ld, int lane) {
  int m = lane & 15, hf = lane >> 4;
  const _Float16* p = s + (size_t)m * ld + hf * 8;
  v8h lo = *(const v8h*)p;
  v8h hi = *(const v8h*)(p + 16);
  v16h f;
#pragma unroll
  for (int i = 0; i < 8; ++i) { f[i] = lo[i]; f[8 + i] = hi[i]; }
  return f;
}

// Same A layout but fp32 source (converted on the fly): 2x (two float4 loads).
__device__ inline v16h frag_a_f32_vec(const float* s, int ld, int lane) {
  int m = lane & 15, hf = lane >> 4;
  const float* p = s + (size_t)m * ld + hf * 8;
  v4f a0 = *(const v4f*)p,        a1 = *(const v4f*)(p + 4);
  v4f b0 = *(const v4f*)(p + 16), b1 = *(const v4f*)(p + 20);
  v16h f;
#pragma unroll
  for (int i = 0; i < 4; ++i) {
    f[i]      = (_Float16)a0[i];
    f[4 + i]  = (_Float16)a1[i];
    f[8 + i]  = (_Float16)b0[i];
    f[12 + i] = (_Float16)b1[i];
  }
  return f;
}

// B-fragment (32x16) from N-major (transposed) source: element(k,n)=s[n*ld+k].
// Per lane (n = lane&15) the 16 K-values are two 8-half runs at k = hf*8 and k = 16+hf*8.
__device__ inline v16h frag_bT_f16_vec(const _Float16* s, int ld, int lane) {
  int n = lane & 15, hf = lane >> 4;
  const _Float16* p = s + (size_t)n * ld + hf * 8;
  v8h lo = *(const v8h*)p;
  v8h hi = *(const v8h*)(p + 16);
  v16h f;
#pragma unroll
  for (int i = 0; i < 8; ++i) { f[i] = lo[i]; f[8 + i] = hi[i]; }
  return f;
}

__device__ inline void atomicMaxF(float* addr, float val) {
  if (val >= 0.0f) atomicMax((int*)addr, __float_as_int(val));
  else             atomicMin((unsigned int*)addr, __float_as_uint(val));
}

// ------------------------------------------------------------ init scratch
__global__ void k_init(float* colmax, float* colsum, float* olacc) {
  int stride = gridDim.x * blockDim.x;
  for (int i = blockIdx.x * blockDim.x + threadIdx.x; i < B_ * H_ * TL_; i += stride) {
    colmax[i] = -__builtin_inff();
    colsum[i] = 0.0f;
  }
  for (int i = blockIdx.x * blockDim.x + threadIdx.x; i < B_ * H_ * TL_ * D_; i += stride)
    olacc[i] = 0.0f;
}

// ---------------- one-time weight prep: fp32 [K,N] -> fp16 transposed [N,K]
__global__ void k_prep_w(const float* __restrict__ W, _Float16* __restrict__ WT) {
  int i = blockIdx.x * blockDim.x + threadIdx.x;   // EMB*EMB
  int n = i & 255, k = i >> 8;
  WT[(size_t)n * EMB_ + k] = (_Float16)W[(size_t)k * EMB_ + n];
}

// ------------------------------------- GEMM: fp32 A [M,256] @ W (fp16, N-major)
// -> fp16 C = (A@W + bias)*scale.  tflag=1: write head-transposed [b,h,d,token].
__global__ __launch_bounds__(128)
void k_gemm_f32_f16(const float* __restrict__ A, const _Float16* __restrict__ WT,
                    const float* __restrict__ bias, _Float16* __restrict__ C,
                    float scale, int ntok, int tflag) {
  int lane = threadIdx.x & 31, w = threadIdx.x >> 5;
  int row0 = blockIdx.x * 64 + w * 16;
  int col0 = blockIdx.y * 64;
  v8f acc[4] = {};
  for (int ks = 0; ks < 8; ++ks) {
    v16h a = frag_a_f32_vec(A + (size_t)row0 * EMB_ + ks * 32, EMB_, lane);
    if (ks < 7)
      __builtin_prefetch(A + (size_t)row0 * EMB_ + (ks + 1) * 32 +
                         (size_t)(lane & 15) * EMB_ + (lane >> 4) * 8, 0, 1);
#pragma unroll
    for (int j = 0; j < 4; ++j) {
      v16h b = frag_bT_f16_vec(WT + (size_t)(col0 + j * 16) * EMB_ + ks * 32, EMB_, lane);
      acc[j] = wmma_f16(a, b, acc[j]);
    }
  }
  int n = lane & 15, hf = lane >> 4;
#pragma unroll
  for (int j = 0; j < 4; ++j) {
    int col = col0 + j * 16 + n;
    float bv = bias[col];
#pragma unroll
    for (int i = 0; i < 8; ++i) {
      int r = row0 + i + 8 * hf;
      float val = (acc[j][i] + bv) * scale;
      if (!tflag) {
        C[(size_t)r * EMB_ + col] = (_Float16)val;
      } else {
        int bb = r / ntok, token = r - bb * ntok;
        int hh = col >> 5, dd = col & 31;
        C[(((size_t)bb * H_ + hh) * D_ + dd) * ntok + token] = (_Float16)val;
      }
    }
  }
}

// ------------------------------------- GEMM: fp16 A [M,256] @ W (fp16, N-major)
// -> fp32 C = A@W + bias   (final output projections)
__global__ __launch_bounds__(128)
void k_gemm_f16_f32(const _Float16* __restrict__ A, const _Float16* __restrict__ WT,
                    const float* __restrict__ bias, float* __restrict__ C) {
  int lane = threadIdx.x & 31, w = threadIdx.x >> 5;
  int row0 = blockIdx.x * 64 + w * 16;
  int col0 = blockIdx.y * 64;
  v8f acc[4] = {};
  for (int ks = 0; ks < 8; ++ks) {
    v16h a = frag_a_f16_vec(A + (size_t)row0 * EMB_ + ks * 32, EMB_, lane);
    if (ks < 7)
      __builtin_prefetch(A + (size_t)row0 * EMB_ + (ks + 1) * 32 +
                         (size_t)(lane & 15) * EMB_ + (lane >> 4) * 8, 0, 1);
#pragma unroll
    for (int j = 0; j < 4; ++j) {
      v16h b = frag_bT_f16_vec(WT + (size_t)(col0 + j * 16) * EMB_ + ks * 32, EMB_, lane);
      acc[j] = wmma_f16(a, b, acc[j]);
    }
  }
  int n = lane & 15, hf = lane >> 4;
#pragma unroll
  for (int j = 0; j < 4; ++j) {
    int col = col0 + j * 16 + n;
    float bv = bias[col];
#pragma unroll
    for (int i = 0; i < 8; ++i) {
      int r = row0 + i + 8 * hf;
      C[(size_t)r * EMB_ + col] = acc[j][i] + bv;
    }
  }
}

// Compute clamped score tile S[32][256] for (b,h, tv0..tv0+31) into LDS.
// wave w: row-group w>>1 (16 rows), column half w&1 (128 cols). K = D = 32.
__device__ inline void compute_S(const _Float16* q16, const _Float16* k16,
                                 int b, int h, int tv0, int lane, int w,
                                 float S[32][260]) {
  int rg = w >> 1, ch = w & 1;
  int n = lane & 15, hf = lane >> 4;
  v16h a = frag_a_f16_vec(q16 + (size_t)(b * TV_ + tv0 + rg * 16) * EMB_ + h * D_,
                          EMB_, lane);
  for (int j = 0; j < 8; ++j) {
    int t = ch * 8 + j;
    // k_head^T fragment: element(k,n) = k16[(tok=t*16+n)*EMB + h*32 + k]
    v16h bf = frag_bT_f16_vec(k16 + (size_t)(b * TL_ + t * 16) * EMB_ + h * D_,
                              EMB_, lane);
    v8f acc = {};
    acc = wmma_f16(a, bf, acc);
#pragma unroll
    for (int i = 0; i < 8; ++i) {
      float s = fminf(fmaxf(acc[i], -MAXC), MAXC);
      S[rg * 16 + i + 8 * hf][t * 16 + n] = s;
    }
  }
}

// ---------------- vision-direction softmax + out_v heads + column max stats
__global__ __launch_bounds__(128)
void k_attn_v(const _Float16* __restrict__ q16, const _Float16* __restrict__ k16,
              const _Float16* __restrict__ vlT, const int* __restrict__ mask_l,
              const unsigned char* __restrict__ mask_v,
              _Float16* __restrict__ ovh, float* __restrict__ colmax) {
  __shared__ float    S[32][260];
  __shared__ _Float16 P[32][264];
  int lane = threadIdx.x & 31, w = threadIdx.x >> 5, tid = threadIdx.x;
  int bh = blockIdx.y, b = bh >> 3, h = bh & 7;
  int tv0 = blockIdx.x * 32;
  int n = lane & 15, hf = lane >> 4;

  compute_S(q16, k16, b, h, tv0, lane, w, S);
  __syncthreads();

  // per-lang-token (column) max over this tv chunk -> global atomic max
  for (int c = tid; c < TL_; c += 128) {
    float cm = -__builtin_inff();
    for (int r = 0; r < 32; ++r)
      if (!mask_v[b * TV_ + tv0 + r]) cm = fmaxf(cm, S[r][c]);
    atomicMaxF(&colmax[bh * TL_ + c], cm);
  }

  // row softmax over lang tokens (additive lang mask)
  if (tid < 32) {
    int r = tid;
    float rmax = -__builtin_inff();
    for (int c = 0; c < TL_; ++c) {
      float s = S[r][c] + (mask_l[b * TL_ + c] ? 0.0f : NEGM);
      rmax = fmaxf(rmax, s);
    }
    float rsum = 0.0f;
    for (int c = 0; c < TL_; ++c) {
      float s = S[r][c] + (mask_l[b * TL_ + c] ? 0.0f : NEGM);
      rsum += __expf(s - rmax);
    }
    float inv = 1.0f / rsum;
    for (int c = 0; c < TL_; ++c) {
      float s = S[r][c] + (mask_l[b * TL_ + c] ? 0.0f : NEGM);
      P[r][c] = (_Float16)(__expf(s - rmax) * inv);
    }
  }
  __syncthreads();

  // out_v head tile: P[32x256] @ vl_head[256x32]; wave w: rowgrp w>>1, ntile w&1
  {
    int mrg = w >> 1, nt = w & 1;
    v8f acc = {};
    for (int ks = 0; ks < 8; ++ks) {
      v16h pa = frag_a_f16_vec(&P[mrg * 16][0] + ks * 32, 264, lane);
      // vlT layout [b,h,d,tok]: element(k=tok,n=d) = vlT[(bh*32+nt*16+n)*TL + k]
      v16h vb = frag_bT_f16_vec(vlT + ((size_t)bh * D_ + nt * 16) * TL_ + ks * 32,
                                TL_, lane);
      acc = wmma_f16(pa, vb, acc);
    }
#pragma unroll
    for (int i = 0; i < 8; ++i) {
      int r = tv0 + mrg * 16 + i + 8 * hf;
      ovh[(size_t)(b * TV_ + r) * EMB_ + h * D_ + nt * 16 + n] = (_Float16)acc[i];
    }
  }
}

// ---------------- lang-direction: E = exp(S - colmax), colsum, E^T @ vv_head
__global__ __launch_bounds__(128)
void k_attn_l(const _Float16* __restrict__ q16, const _Float16* __restrict__ k16,
              const _Float16* __restrict__ vvT, const unsigned char* __restrict__ mask_v,
              const float* __restrict__ colmax, float* __restrict__ colsum,
              float* __restrict__ olacc) {
  __shared__ float    S[32][260];
  __shared__ _Float16 Et[256][40];   // E transposed: [tl][tv_local], padded
  int lane = threadIdx.x & 31, w = threadIdx.x >> 5, tid = threadIdx.x;
  int bh = blockIdx.y, b = bh >> 3, h = bh & 7;
  int tv0 = blockIdx.x * 32;
  int n = lane & 15, hf = lane >> 4;

  compute_S(q16, k16, b, h, tv0, lane, w, S);
  __syncthreads();

  for (int idx = tid; idx < 32 * TL_; idx += 128) {
    int r = idx >> 8, c = idx & 255;
    float e = 0.0f;
    if (!mask_v[b * TV_ + tv0 + r]) {
      float cm = colmax[bh * TL_ + c];
      if (cm > -__builtin_inff()) e = __expf(S[r][c] - cm);
    }
    Et[c][r] = (_Float16)e;
  }
  __syncthreads();

  // partial column sums (softmax denominators over tv)
  for (int c = tid; c < TL_; c += 128) {
    float s = 0.0f;
    for (int r = 0; r < 32; ++r) s += (float)Et[c][r];
    atomicAdd(&colsum[bh * TL_ + c], s);
  }

  // out_l partial: E^T [256 x 32] @ vv_head [32 x 32]; wave w owns 64 tl rows
  // vvT layout [b,h,d,tok]: element(k=tok,n=d) = vvT[(bh*32+nt*16+n)*TV + tv0+k]
  v16h vb0 = frag_bT_f16_vec(vvT + ((size_t)bh * D_) * TV_ + tv0, TV_, lane);
  v16h vb1 = frag_bT_f16_vec(vvT + ((size_t)bh * D_ + 16) * TV_ + tv0, TV_, lane);
#pragma unroll
  for (int mt = 0; mt < 4; ++mt) {
    int m0 = w * 64 + mt * 16;
    v16h ea = frag_a_f16_vec(&Et[m0][0], 40, lane);
    v8f a0 = {}, a1 = {};
    a0 = wmma_f16(ea, vb0, a0);
    a1 = wmma_f16(ea, vb1, a1);
#pragma unroll
    for (int i = 0; i < 8; ++i) {
      int m = m0 + i + 8 * hf;
      atomicAdd(&olacc[((size_t)bh * TL_ + m) * D_ + n],      a0[i]);
      atomicAdd(&olacc[((size_t)bh * TL_ + m) * D_ + 16 + n], a1[i]);
    }
  }
}

// ---------------- normalize out_l heads, pack to [B*TL, EMB] fp16
__global__ void k_norm_l(const float* __restrict__ olacc, const float* __restrict__ colsum,
                         _Float16* __restrict__ olh) {
  int i = blockIdx.x * blockDim.x + threadIdx.x;
  if (i >= B_ * H_ * TL_ * D_) return;
  int d = i & 31, t = (i >> 5) & 255, bh = i >> 13;
  int b = bh >> 3, h = bh & 7;
  float cs = colsum[bh * TL_ + t];
  float v = (cs > 0.0f) ? olacc[i] / cs : 0.0f;
  olh[(size_t)(b * TL_ + t) * EMB_ + h * D_ + d] = (_Float16)v;
}

// ---------------------------------------------------------------- launcher
extern "C" void kernel_launch(void* const* d_in, const int* in_sizes, int n_in,
                              void* d_out, int out_size, void* d_ws, size_t ws_size,
                              hipStream_t stream) {
  const float* vision = (const float*)d_in[0];
  const float* lang   = (const float*)d_in[1];
  const unsigned char* mask_v = (const unsigned char*)d_in[2]; // bool, all false
  const int*   mask_l = (const int*)d_in[3];                   // int, all ones
  const float* Wq  = (const float*)d_in[4];  const float* bq  = (const float*)d_in[5];
  const float* Wk  = (const float*)d_in[6];  const float* bk  = (const float*)d_in[7];
  const float* Wvv = (const float*)d_in[8];  const float* bvv = (const float*)d_in[9];
  const float* Wvl = (const float*)d_in[10]; const float* bvl = (const float*)d_in[11];
  const float* Wov = (const float*)d_in[12]; const float* bov = (const float*)d_in[13];
  const float* Wol = (const float*)d_in[14]; const float* bol = (const float*)d_in[15];

  float* out_v = (float*)d_out;
  float* out_l = out_v + (size_t)B_ * TV_ * EMB_;

  char* ws = (char*)d_ws;
  size_t off = 0;
  auto alloc = [&](size_t bytes) -> void* {
    void* p = ws + off;
    off += (bytes + 255) & ~(size_t)255;
    return p;
  };
  _Float16* q16  = (_Float16*)alloc((size_t)B_ * TV_ * EMB_ * 2);
  _Float16* k16  = (_Float16*)alloc((size_t)B_ * TL_ * EMB_ * 2);
  _Float16* vvT  = (_Float16*)alloc((size_t)B_ * TV_ * EMB_ * 2); // [b,h,d,tok]
  _Float16* vlT  = (_Float16*)alloc((size_t)B_ * TL_ * EMB_ * 2); // [b,h,d,tok]
  _Float16* ovh  = (_Float16*)alloc((size_t)B_ * TV_ * EMB_ * 2);
  _Float16* olh  = (_Float16*)alloc((size_t)B_ * TL_ * EMB_ * 2);
  float* colmax  = (float*)alloc((size_t)B_ * H_ * TL_ * 4);
  float* colsum  = (float*)alloc((size_t)B_ * H_ * TL_ * 4);
  float* olacc   = (float*)alloc((size_t)B_ * H_ * TL_ * D_ * 4);
  _Float16* WqT  = (_Float16*)alloc((size_t)EMB_ * EMB_ * 2);
  _Float16* WkT  = (_Float16*)alloc((size_t)EMB_ * EMB_ * 2);
  _Float16* WvvT = (_Float16*)alloc((size_t)EMB_ * EMB_ * 2);
  _Float16* WvlT = (_Float16*)alloc((size_t)EMB_ * EMB_ * 2);
  _Float16* WovT = (_Float16*)alloc((size_t)EMB_ * EMB_ * 2);
  _Float16* WolT = (_Float16*)alloc((size_t)EMB_ * EMB_ * 2);

  const float scl = 1.0f / sqrtf((float)D_);
  dim3 blk(128);

  k_init<<<dim3(64), dim3(256), 0, stream>>>(colmax, colsum, olacc);

  dim3 pw(EMB_ * EMB_ / 256);
  k_prep_w<<<pw, dim3(256), 0, stream>>>(Wq,  WqT);
  k_prep_w<<<pw, dim3(256), 0, stream>>>(Wk,  WkT);
  k_prep_w<<<pw, dim3(256), 0, stream>>>(Wvv, WvvT);
  k_prep_w<<<pw, dim3(256), 0, stream>>>(Wvl, WvlT);
  k_prep_w<<<pw, dim3(256), 0, stream>>>(Wov, WovT);
  k_prep_w<<<pw, dim3(256), 0, stream>>>(Wol, WolT);

  k_gemm_f32_f16<<<dim3((B_ * TV_) / 64, 4), blk, 0, stream>>>(vision, WqT,  bq,  q16, scl,  TV_, 0);
  k_gemm_f32_f16<<<dim3((B_ * TL_) / 64, 4), blk, 0, stream>>>(lang,   WkT,  bk,  k16, 1.0f, TL_, 0);
  k_gemm_f32_f16<<<dim3((B_ * TV_) / 64, 4), blk, 0, stream>>>(vision, WvvT, bvv, vvT, 1.0f, TV_, 1);
  k_gemm_f32_f16<<<dim3((B_ * TL_) / 64, 4), blk, 0, stream>>>(lang,   WvlT, bvl, vlT, 1.0f, TL_, 1);

  k_attn_v<<<dim3(TV_ / 32, B_ * H_), blk, 0, stream>>>(q16, k16, vlT, mask_l, mask_v,
                                                        ovh, colmax);
  k_attn_l<<<dim3(TV_ / 32, B_ * H_), blk, 0, stream>>>(q16, k16, vvT, mask_v, colmax,
                                                        colsum, olacc);
  k_norm_l<<<dim3((B_ * H_ * TL_ * D_) / 256), dim3(256), 0, stream>>>(olacc, colsum, olh);

  k_gemm_f16_f32<<<dim3((B_ * TV_) / 64, 4), blk, 0, stream>>>(ovh, WovT, bov, out_v);
  k_gemm_f16_f32<<<dim3((B_ * TL_) / 64, 4), blk, 0, stream>>>(olh, WolT, bol, out_l);
}
